// DGCNN_29626684407868
// MI455X (gfx1250) — matrix operand order repaired and checked by reading, not compiled, not run on hardware
//
#include <hip/hip_runtime.h>
#include <hip/hip_bf16.h>

// ---------------- CDNA5 WMMA types ----------------
typedef __attribute__((ext_vector_type(16))) _Float16 v16h;
typedef __attribute__((ext_vector_type(8)))  float    v8f;

#define BATCH 8
#define NPTS  2048
#define KNN   20
#define KPAD  32
#define SLOPE 0.2f

extern __shared__ unsigned char dynsmem[];

__device__ __forceinline__ float lrelu(float v) { return v > 0.f ? v : SLOPE * v; }

// ---------------------------------------------------------------------------
// Transpose input x [B,3,N] (channel-major) -> xinT [B,N,4] (point-major, padded
// with a zero 4th channel so KNN can run with Cin=4 branch-free).
// ---------------------------------------------------------------------------
__global__ void transpose_in_kernel(const float* __restrict__ x, float* __restrict__ xt) {
    int i = blockIdx.x * blockDim.x + threadIdx.x;   // over B*N
    if (i >= BATCH * NPTS) return;
    int b = i / NPTS, n = i % NPTS;
    const float* xb = x + (size_t)b * 3 * NPTS;
    float* o = xt + (size_t)i * 4;
    o[0] = xb[0 * NPTS + n];
    o[1] = xb[1 * NPTS + n];
    o[2] = xb[2 * NPTS + n];
    o[3] = 0.f;
}

// ---------------------------------------------------------------------------
// Convert + zero-pad weight [Cout, twoC] f32 -> [Cout, twoCp] f16
// ---------------------------------------------------------------------------
__global__ void wpad_kernel(const float* __restrict__ W, _Float16* __restrict__ Wh,
                            int Cout, int twoC, int twoCp) {
    int i = blockIdx.x * blockDim.x + threadIdx.x;
    if (i >= Cout * twoCp) return;
    int o = i / twoCp, k = i % twoCp;
    Wh[i] = (k < twoC) ? (_Float16)W[(size_t)o * twoC + k] : (_Float16)0.f;
}

// ---------------------------------------------------------------------------
// KNN: one thread per query (64 queries/block); candidate chunks of 32 rows are
// staged into LDS with CDNA5 async global->LDS copies (ASYNCcnt), then dot
// products run register-tiled (query chunk of 4 in VGPRs, 32 candidate
// accumulators in VGPRs, ds_load_b128-able candidate reads).
// Cin must be a multiple of 4. Output idx padded to KPAD with the self index.
// ---------------------------------------------------------------------------
__global__ void knn_kernel(const float* __restrict__ feat, int stride, int coff, int Cin,
                           int* __restrict__ idxout) {
    float* qs = (float*)dynsmem;            // [64][Cin]
    float* cs = qs + 64 * Cin;              // [32][Cin]
    float* cn = cs + 32 * Cin;              // [32]
    int t   = threadIdx.x;                  // 0..63
    int blk = blockIdx.x;
    int b   = blk / (NPTS / 64);
    int q   = (blk % (NPTS / 64)) * 64 + t;
    const float* fb = feat + (size_t)b * NPTS * stride + coff;
    const unsigned cs_lds = (unsigned)(size_t)cs;   // LDS byte address of candidate tile
    const int cvec = Cin >> 2;                      // 16B chunks per row

    for (int c = 0; c < Cin; ++c) qs[t * Cin + c] = fb[(size_t)q * stride + c];
    float qq = 0.f;
    for (int c = 0; c < Cin; ++c) { float v = qs[t * Cin + c]; qq += v * v; }

    float kd[KNN]; int ki[KNN];
    for (int r = 0; r < KNN; ++r) { kd[r] = 3.4e38f; ki[r] = q; }

    for (int j0 = 0; j0 < NPTS; j0 += 32) {
        __syncthreads();                    // previous-iteration readers done
        // ---- async global->LDS staging of 32 candidate rows (b128 granules) ----
        for (int e = t; e < 32 * cvec; e += 64) {
            int row = e / cvec, c4 = e % cvec;
            const float* gp = fb + (size_t)(j0 + row) * stride + c4 * 4;
            unsigned ldst = cs_lds + (unsigned)(row * Cin + c4 * 4) * 4u;
            asm volatile("global_load_async_to_lds_b128 %0, %1, off"
                         :: "v"(ldst), "v"(gp) : "memory");
        }
        asm volatile("s_wait_asynccnt 0x0" ::: "memory");
        __syncthreads();                    // all waves' async copies visible
        if (t < 32) {                       // candidate squared norms
            float s = 0.f;
            for (int c = 0; c < Cin; ++c) { float v = cs[t * Cin + c]; s += v * v; }
            cn[t] = s;
        }
        __syncthreads();

        // ---- register-tiled dot products: 32 accumulators, query chunk in VGPRs ----
        float acc[32];
        #pragma unroll
        for (int jj = 0; jj < 32; ++jj) acc[jj] = 0.f;
        for (int c0 = 0; c0 < Cin; c0 += 4) {
            float q0 = qs[t * Cin + c0 + 0];
            float q1 = qs[t * Cin + c0 + 1];
            float q2 = qs[t * Cin + c0 + 2];
            float q3 = qs[t * Cin + c0 + 3];
            #pragma unroll
            for (int jj = 0; jj < 32; ++jj) {
                const float* cr = &cs[jj * Cin + c0];
                acc[jj] += q0 * cr[0] + q1 * cr[1] + q2 * cr[2] + q3 * cr[3];
            }
        }
        for (int jj = 0; jj < 32; ++jj) {
            float d2 = qq + cn[jj] - 2.f * acc[jj];
            if (d2 < kd[KNN - 1]) {
                int p = KNN - 1;
                while (p > 0 && kd[p - 1] > d2) { kd[p] = kd[p - 1]; ki[p] = ki[p - 1]; --p; }
                kd[p] = d2; ki[p] = j0 + jj;
            }
        }
    }
    int* op = idxout + ((size_t)b * NPTS + q) * KPAD;
    for (int r = 0; r < KNN; ++r) op[r] = ki[r];
    for (int r = KNN; r < KPAD; ++r) op[r] = q;   // self-pad: edge (0, ctr) == self edge
}

// ---------------------------------------------------------------------------
// EdgeConv: per point, build [KPAD, 2C] edge features (nbr-ctr | ctr) in LDS
// as f16, GEMM against W^T via v_wmma_f32_16x16x32_f16, leakyReLU, max over
// the M (=K neighbors) dimension of the WMMA accumulator.
// One wave per point, 4 waves per block.
// ---------------------------------------------------------------------------
__global__ void edge_kernel(const float* __restrict__ feat, int in_stride, int in_coff, int Cin,
                            const int* __restrict__ idx, const _Float16* __restrict__ Wh, int Cout,
                            float* __restrict__ outf32, _Float16* __restrict__ outf16,
                            int out_stride, int out_coff) {
    const int twoC  = 2 * Cin;
    const int twoCp = (twoC + 31) & ~31;
    int lane = threadIdx.x & 31;
    int wave = threadIdx.x >> 5;
    _Float16* A = (_Float16*)dynsmem + (size_t)wave * KPAD * twoCp;

    int gw = blockIdx.x * (blockDim.x >> 5) + wave;   // point id over B*N
    int b = gw / NPTS, n = gw % NPTS;

    const float* fb = feat + (size_t)b * NPTS * in_stride + in_coff;
    const int*   ip = idx + ((size_t)b * NPTS + n) * KPAD;

    // gather: lane r fills edge-feature row r
    int r  = lane;
    int nb = ip[r];
    for (int c = 0; c < Cin; ++c) {
        float cv = fb[(size_t)n * in_stride + c];
        float nv = fb[(size_t)nb * in_stride + c];
        A[r * twoCp + c]       = (_Float16)(nv - cv);
        A[r * twoCp + Cin + c] = (_Float16)cv;
    }
    for (int c = twoC; c < twoCp; ++c) A[r * twoCp + c] = (_Float16)0.f;
    __syncthreads();

    // WMMA fragment addressing (ISA 7.12.2, 16-bit layouts)
    int m     = lane & 15;
    int abase = (lane < 16) ? 0 : 8;    // A: K-halves split across lane groups
    int kb    = (lane < 16) ? 0 : 8;    // B: K rows split across lane groups

    for (int o0 = 0; o0 < Cout; o0 += 16) {
        v8f acc0 = {}; v8f acc1 = {};
        for (int kt = 0; kt < twoCp; kt += 32) {
            v16h af0, af1, bf;
            const _Float16* a0 = &A[m * twoCp + kt];
            const _Float16* a1 = &A[(16 + m) * twoCp + kt];
            #pragma unroll
            for (int i = 0; i < 8; ++i) {
                af0[i]     = a0[abase + i];
                af0[8 + i] = a0[16 + abase + i];
                af1[i]     = a1[abase + i];
                af1[8 + i] = a1[16 + abase + i];
            }
            const _Float16* wp = Wh + (size_t)(o0 + m) * twoCp + kt + kb; // B[k][o]=W[o][k]
            #pragma unroll
            for (int i = 0; i < 16; ++i) bf[i] = wp[i < 8 ? i : i + 8];
            acc0 = __builtin_amdgcn_wmma_f32_16x16x32_f16(false, af0, false, bf,
                                                          (short)0, acc0, false, false);
            acc1 = __builtin_amdgcn_wmma_f32_16x16x32_f16(false, af1, false, bf,
                                                          (short)0, acc1, false, false);
        }
        // leakyReLU + max over M (= padded neighbor dim)
        float mx = -3.4e38f;
        #pragma unroll
        for (int i = 0; i < 8; ++i)
            mx = fmaxf(mx, fmaxf(lrelu(acc0[i]), lrelu(acc1[i])));
        mx = fmaxf(mx, __shfl_xor(mx, 16, 32));   // combine row halves (lanes L <-> L+16)
        if (lane < 16) {
            size_t o = ((size_t)b * NPTS + n) * out_stride + out_coff + o0 + lane;
            outf32[o] = mx;
            outf16[o] = (_Float16)mx;
        }
    }
}

// ---------------------------------------------------------------------------
// Global conv: g = leakyReLU(Wg @ xcat), max over N.
// Block = (b, 16-channel tile); 4 waves stride over n-tiles, WMMA N = points,
// max folded elementwise across n-tiles then lane xor-tree, then LDS combine.
// ---------------------------------------------------------------------------
__global__ void gconv_kernel(const _Float16* __restrict__ xh, const _Float16* __restrict__ Wgh,
                             float* __restrict__ out) {
    __shared__ float red[4][16];
    int lane = threadIdx.x & 31;
    int wave = threadIdx.x >> 5;
    int b  = blockIdx.x >> 6;
    int o0 = (blockIdx.x & 63) * 16;

    int m     = lane & 15;
    int abase = (lane < 16) ? 0 : 8;
    int kb    = (lane < 16) ? 0 : 8;

    v8f mx;
    #pragma unroll
    for (int i = 0; i < 8; ++i) mx[i] = -3.4e38f;

    for (int nt = wave; nt < NPTS / 16; nt += 4) {
        int n = nt * 16 + m;
        const _Float16* brow = xh + ((size_t)b * NPTS + n) * 512;
        __builtin_prefetch(brow + 4 * 16 * 512, 0, 0);  // next n-tile for this wave
        v8f acc = {};
        for (int kt = 0; kt < 512; kt += 32) {
            v16h af, bf;
            const _Float16* ap = Wgh + (size_t)(o0 + m) * 512 + kt;   // A[m=o][k=c]
            #pragma unroll
            for (int i = 0; i < 8; ++i) { af[i] = ap[abase + i]; af[8 + i] = ap[16 + abase + i]; }
            const _Float16* bp = brow + kt + kb;                      // B[k=c][n]
            #pragma unroll
            for (int i = 0; i < 16; ++i) bf[i] = bp[i < 8 ? i : i + 8];
            acc = __builtin_amdgcn_wmma_f32_16x16x32_f16(false, af, false, bf,
                                                         (short)0, acc, false, false);
        }
        #pragma unroll
        for (int i = 0; i < 8; ++i) mx[i] = fmaxf(mx[i], lrelu(acc[i]));
    }
    // reduce max over the 16 columns within each lane half-group
    #pragma unroll
    for (int off = 1; off < 16; off <<= 1)
        #pragma unroll
        for (int i = 0; i < 8; ++i) mx[i] = fmaxf(mx[i], __shfl_xor(mx[i], off, 32));
    if (m == 0) {
        int rb = (lane < 16) ? 0 : 8;   // VGPR i -> rows i / i+8
        #pragma unroll
        for (int i = 0; i < 8; ++i) red[wave][rb + i] = mx[i];
    }
    __syncthreads();
    if (threadIdx.x < 16) {
        float v = red[0][threadIdx.x];
        for (int w = 1; w < 4; ++w) v = fmaxf(v, red[w][threadIdx.x]);
        out[(size_t)b * 1024 + o0 + threadIdx.x] = v;
    }
}

// ---------------------------------------------------------------------------
extern "C" void kernel_launch(void* const* d_in, const int* in_sizes, int n_in,
                              void* d_out, int out_size, void* d_ws, size_t ws_size,
                              hipStream_t stream) {
    const float* x  = (const float*)d_in[0];   // [8,3,2048]
    const float* W1 = (const float*)d_in[1];   // [64,6]
    const float* W2 = (const float*)d_in[2];   // [64,128]
    const float* W3 = (const float*)d_in[3];   // [128,128]
    const float* W4 = (const float*)d_in[4];   // [256,256]
    const float* Wg = (const float*)d_in[5];   // [1024,512]
    float* out = (float*)d_out;                // [8,1024]

    // workspace carve-up (256B aligned regions)
    size_t off = 0;
    auto carve = [&](size_t bytes) { size_t o = off; off = (off + bytes + 255) & ~(size_t)255; return o; };
    char* ws = (char*)d_ws;
    float*     xinT = (float*)    (ws + carve((size_t)BATCH * NPTS * 4 * 4));
    float*     xc32 = (float*)    (ws + carve((size_t)BATCH * NPTS * 512 * 4));
    _Float16*  xc16 = (_Float16*) (ws + carve((size_t)BATCH * NPTS * 512 * 2));
    int*       idx  = (int*)      (ws + carve((size_t)BATCH * NPTS * KPAD * 4));
    _Float16*  W1h  = (_Float16*) (ws + carve((size_t)64 * 32 * 2));
    _Float16*  W2h  = (_Float16*) (ws + carve((size_t)64 * 128 * 2));
    _Float16*  W3h  = (_Float16*) (ws + carve((size_t)128 * 128 * 2));
    _Float16*  W4h  = (_Float16*) (ws + carve((size_t)256 * 256 * 2));
    _Float16*  Wgh  = (_Float16*) (ws + carve((size_t)1024 * 512 * 2));

    const int BN = BATCH * NPTS;

    transpose_in_kernel<<<(BN + 255) / 256, 256, 0, stream>>>(x, xinT);
    wpad_kernel<<<(64 * 32 + 255) / 256, 256, 0, stream>>>(W1, W1h, 64, 6, 32);
    wpad_kernel<<<(64 * 128 + 255) / 256, 256, 0, stream>>>(W2, W2h, 64, 128, 128);
    wpad_kernel<<<(128 * 128 + 255) / 256, 256, 0, stream>>>(W3, W3h, 128, 128, 128);
    wpad_kernel<<<(256 * 256 + 255) / 256, 256, 0, stream>>>(W4, W4h, 256, 256, 256);
    wpad_kernel<<<(1024 * 512 + 255) / 256, 256, 0, stream>>>(Wg, Wgh, 1024, 512, 512);

    auto knn_lds  = [](int Cin) { return (size_t)(64 * Cin + 32 * Cin + 32) * 4; };
    auto edge_lds = [](int Cin) { int t = 2 * Cin; t = (t + 31) & ~31; return (size_t)4 * KPAD * t * 2; };
    const int knn_grid  = BN / 64;
    const int edge_grid = BN / 4;

    // block1: in xinT (stride 4, KNN Cin=4 w/ zero pad channel, conv Cin=3) -> xcat [0,64)
    knn_kernel <<<knn_grid, 64, knn_lds(4), stream>>>(xinT, 4, 0, 4, idx);
    edge_kernel<<<edge_grid, 128, edge_lds(3), stream>>>(xinT, 4, 0, 3, idx, W1h, 64,
                                                         xc32, xc16, 512, 0);
    // block2: in xcat[0,64) -> xcat cols [64,128)
    knn_kernel <<<knn_grid, 64, knn_lds(64), stream>>>(xc32, 512, 0, 64, idx);
    edge_kernel<<<edge_grid, 128, edge_lds(64), stream>>>(xc32, 512, 0, 64, idx, W2h, 64,
                                                          xc32, xc16, 512, 64);
    // block3: in xcat[64,128) -> xcat cols [128,256)
    knn_kernel <<<knn_grid, 64, knn_lds(64), stream>>>(xc32, 512, 64, 64, idx);
    edge_kernel<<<edge_grid, 128, edge_lds(64), stream>>>(xc32, 512, 64, 64, idx, W3h, 128,
                                                          xc32, xc16, 512, 128);
    // block4: in xcat[128,256) -> xcat cols [256,512)
    knn_kernel <<<knn_grid, 64, knn_lds(128), stream>>>(xc32, 512, 128, 128, idx);
    edge_kernel<<<edge_grid, 128, edge_lds(128), stream>>>(xc32, 512, 128, 128, idx, W4h, 256,
                                                           xc32, xc16, 512, 256);
    // global conv + max over N
    gconv_kernel<<<BATCH * 64, 128, 0, stream>>>(xc16, Wgh, out);
}
